// CoNHD_ADMM_87282325389901
// MI455X (gfx1250) — compile-verified, matrix operand order checked by reading, not run on hardware
//
#include <hip/hip_runtime.h>

typedef __attribute__((ext_vector_type(16))) _Float16 v16h;
typedef __attribute__((ext_vector_type(8)))  _Float16 v8h;
typedef __attribute__((ext_vector_type(8)))  float    v8f;
typedef __attribute__((ext_vector_type(4)))  float    v4f;

#define D_DIM 64
#define E_CNT 1000000
#define TILES_PER_STREAM (E_CNT / 16)      // 62500 exact, no partial tiles
#define N_TILES (2 * TILES_PER_STREAM)     // 125000

__global__ __launch_bounds__(256) void conhd_gather_gemm_relu(
    const float* __restrict__ vfeat,   // [50000, 64] f32
    const int*   __restrict__ in_src,  // [1e6]
    const int*   __restrict__ con_dst, // [1e6]
    const float* __restrict__ W,       // [64, 64] f32, W[k][n]
    const float* __restrict__ b,       // [64] f32
    float*       __restrict__ out)     // [2, 1e6, 64] f32
{
    // ---- Stage W^T into LDS as f16 once per block (8 KB) ----
    __shared__ __align__(16) _Float16 Wt[D_DIM * D_DIM];   // Wt[n][k]
    for (int i = threadIdx.x; i < D_DIM * D_DIM; i += blockDim.x) {
        const int n = i >> 6;
        const int k = i & 63;
        Wt[n * D_DIM + k] = (_Float16)W[k * D_DIM + n];
    }
    __syncthreads();

    const int lane  = threadIdx.x & 31;
    const int nCol  = lane & 15;   // B/C column within a 16-wide tile; A row index
    const int kHalf = lane >> 4;   // which K sub-chunk this half-wave holds
    const int klo   = kHalf * 8;

    // ---- Load all 8 B fragments into registers once per wave ----
    // 16-bit B fragment mirrors A layout: lane holds column n of W, chunks
    // [kb*32+klo .. +8) and [kb*32+klo+16 .. +8).
    v16h Bfrag[4][2];
#pragma unroll
    for (int nt = 0; nt < 4; ++nt) {
        const int n = nt * 16 + nCol;
        const _Float16* wp = &Wt[n * D_DIM];
#pragma unroll
        for (int kb = 0; kb < 2; ++kb) {
            const int k0 = kb * 32 + klo;
            v8h lo = *(const v8h*)(wp + k0);        // ds_load_b128
            v8h hi = *(const v8h*)(wp + k0 + 16);   // ds_load_b128
            v16h f;
#pragma unroll
            for (int i = 0; i < 8; ++i) { f[i] = lo[i]; f[8 + i] = hi[i]; }
            Bfrag[nt][kb] = f;
        }
    }

    // Per-lane bias for each N tile (C layout: lane's column = nt*16 + nCol)
    float bias[4];
#pragma unroll
    for (int nt = 0; nt < 4; ++nt) bias[nt] = b[nt * 16 + nCol];

    const int wavesPerBlock = blockDim.x >> 5;
    const int waveId = blockIdx.x * wavesPerBlock + (threadIdx.x >> 5);
    const int nWaves = gridDim.x * wavesPerBlock;

    // ---- Grid-stride loop over 16-edge output tiles (wave-uniform) ----
    for (int tile = waveId; tile < N_TILES; tile += nWaves) {
        const int  stream = (tile >= TILES_PER_STREAM) ? 1 : 0;
        const int  eBase  = (stream ? (tile - TILES_PER_STREAM) : tile) * 16;
        const int* __restrict__ idx = stream ? con_dst : in_src;

        // Both half-waves gather the same 16 rows (coalesced 64B index read),
        // but load different K chunks per the 16-bit A-matrix VGPR layout.
        const int row = idx[eBase + nCol];
        const float* __restrict__ src = vfeat + (size_t)row * D_DIM;

        v16h A[2];
#pragma unroll
        for (int kb = 0; kb < 2; ++kb) {
            const int k0 = kb * 32 + klo;
            v4f f0 = *(const v4f*)(src + k0);        // global_load_b128 (L2-resident)
            v4f f1 = *(const v4f*)(src + k0 + 4);
            v4f f2 = *(const v4f*)(src + k0 + 16);
            v4f f3 = *(const v4f*)(src + k0 + 20);
            v16h a;
#pragma unroll
            for (int i = 0; i < 4; ++i) {
                a[i]      = (_Float16)f0[i];
                a[4 + i]  = (_Float16)f1[i];
                a[8 + i]  = (_Float16)f2[i];
                a[12 + i] = (_Float16)f3[i];
            }
            A[kb] = a;
        }

        float* __restrict__ obase =
            out + ((size_t)stream * E_CNT + (size_t)eBase) * D_DIM;

#pragma unroll
        for (int nt = 0; nt < 4; ++nt) {
            v8f c = {};
            // K = 0..31 then K = 32..63   (8 args: neg_a, A, neg_b, B, c_mod, C, reuse_a, reuse_b)
            c = __builtin_amdgcn_wmma_f32_16x16x32_f16(
                    false, A[0], false, Bfrag[nt][0], (short)0, c, false, false);
            c = __builtin_amdgcn_wmma_f32_16x16x32_f16(
                    false, A[1], false, Bfrag[nt][1], (short)0, c, false, false);

            // C VGPR r holds row m = r + kHalf*8, column nt*16 + nCol.
            // Bias + ReLU in registers, streaming (non-temporal) stores:
            // output is 512MB write-once — keep it out of L2 so vfeat stays hot.
#pragma unroll
            for (int r = 0; r < 8; ++r) {
                float v = c[r] + bias[nt];
                v = v > 0.0f ? v : 0.0f;
                __builtin_nontemporal_store(
                    v, obase + (size_t)(r + kHalf * 8) * D_DIM + nt * 16 + nCol);
            }
        }
    }
}

extern "C" void kernel_launch(void* const* d_in, const int* in_sizes, int n_in,
                              void* d_out, int out_size, void* d_ws, size_t ws_size,
                              hipStream_t stream) {
    const float* vfeat   = (const float*)d_in[0];
    const int*   in_src  = (const int*)  d_in[1];
    const int*   con_dst = (const int*)  d_in[2];
    const float* W_in    = (const float*)d_in[3];
    const float* b_in    = (const float*)d_in[4];
    float* out = (float*)d_out;

    // 2048 blocks * 8 waves = 16384 waves over 125000 tiles (~7.6 tiles/wave):
    // plenty of latency hiding for a store-BW-bound kernel, while amortizing
    // the per-wave B-fragment register load.
    dim3 grid(2048), block(256);
    hipLaunchKernelGGL(conhd_gather_gemm_relu, grid, block, 0, stream,
                       vfeat, in_src, con_dst, W_in, b_in, out);
}
